// GumbelSoftmaxTokenizer_9483287789753
// MI455X (gfx1250) — matrix operand order repaired and checked by reading, not compiled, not run on hardware
//
#include <hip/hip_runtime.h>
#include <hip/hip_bf16.h>
#include <stdint.h>

// ---------------------------------------------------------------------------
// Types for CDNA5 WMMA (wave32): 16x16x32 f16 -> f32 accumulate
// ---------------------------------------------------------------------------
typedef _Float16 half_t;
typedef __attribute__((ext_vector_type(8)))  _Float16 v8h;
typedef __attribute__((ext_vector_type(16))) _Float16 v16h;
typedef __attribute__((ext_vector_type(8)))  float    v8f;

// TDM descriptor group types (ROCm 7.2 probe: (u32x4, i32x8, i32x4, i32x4, i32))
typedef __attribute__((ext_vector_type(4))) unsigned int v4u;
typedef __attribute__((ext_vector_type(8))) int          v8i_t;
typedef __attribute__((ext_vector_type(4))) int          v4i_t;

union Frag16 { v16h v; v8h h[2]; };

#define BM 128
#define BN 128
#define BK 32
#define LDP 40   // padded row stride (halves): 80B, keeps 16B alignment

// ---------------------------------------------------------------------------
// TDM: DMA one 128-row x 32-col f16 tile (row stride = kstride elems) from
// global into LDS at byte offset lds_off, emitting padded rows that match the
// sA/sB layout: pad_interval = 16 DWORDs (one 64B row), pad_amount = 4 DWORDs
// (16B -> 40-half padded stride). Tracked by TENSORcnt.
// ---------------------------------------------------------------------------
__device__ __forceinline__ void tdm_load_tile_128x32(const half_t* gsrc,
                                                     unsigned int lds_off,
                                                     int kstride, int rows)
{
  unsigned long long ga = (unsigned long long)(uintptr_t)gsrc;

  // D# group 0: [1:0]=count=1, [63:32]=lds_addr, [120:64]=global_addr, [127:126]=type(2)
  v4u g0;
  g0.x = 1u;
  g0.y = lds_off;
  g0.z = (unsigned int)ga;
  g0.w = ((unsigned int)(ga >> 32) & 0x01FFFFFFu) | (2u << 30);

  // D# group 1 packed via 64-bit quads
  unsigned long long td0 = (unsigned long long)(unsigned int)kstride; // tensor_dim0
  unsigned long long td1 = (unsigned long long)(unsigned int)rows;    // tensor_dim1
  unsigned long long st0 = (unsigned long long)(unsigned int)kstride; // dim0 stride
  // w0: data_size=1(2B)<<16 | pad_enable<<20 | pad_interval=3(16DW)<<22 | pad_amount=3(4DW)<<25
  unsigned long long q0 = ((1ull << 16) | (1ull << 20) | (3ull << 22) | (3ull << 25))
                        | (td0 << 48);                 // tensor_dim0[15:0] @ bits 63:48
  unsigned long long q1 = (td0 >> 16)                  // tensor_dim0[31:16]
                        | (td1 << 16)                  // tensor_dim1 @ bits 111:80
                        | (32ull << 48);               // tile_dim0 = 32 @ bits 127:112
  unsigned long long q2 = 128ull                       // tile_dim1 = 128 @ bits 143:128
                        | (st0 << 32);                 // dim0_stride[31:0] @ bits 191:160
  unsigned long long q3 = (st0 >> 32);                 // dim0_stride[47:32] @ bits 207:192
  v8i_t g1;
  g1[0] = (int)q0; g1[1] = (int)(q0 >> 32);
  g1[2] = (int)q1; g1[3] = (int)(q1 >> 32);
  g1[4] = (int)q2; g1[5] = (int)(q2 >> 32);
  g1[6] = (int)q3; g1[7] = (int)(q3 >> 32);

  v4i_t gz = {0, 0, 0, 0};
#if defined(__clang_major__) && (__clang_major__ >= 23)
  v8i_t gz8 = {0, 0, 0, 0, 0, 0, 0, 0};
  __builtin_amdgcn_tensor_load_to_lds(g0, g1, gz, gz, gz8, 0);
#else
  __builtin_amdgcn_tensor_load_to_lds(g0, g1, gz, gz, 0);
#endif
}

// ---------------------------------------------------------------------------
// One-time converters: weights -> f16 transposed (N x K), activations -> f16
// ---------------------------------------------------------------------------
__global__ void __launch_bounds__(256)
convert_f16_kernel(const float* __restrict__ src, half_t* __restrict__ dst, int n)
{
  int i = blockIdx.x * 256 + threadIdx.x;
  if (i < n) dst[i] = (half_t)src[i];
}

// Wt[n*K + k] = (half)W[k*N + n]
__global__ void __launch_bounds__(256)
transpose_f16_kernel(const float* __restrict__ W, half_t* __restrict__ Wt, int K, int N)
{
  int i = blockIdx.x * 256 + threadIdx.x;
  if (i < K * N) {
    int n = i / K, k = i - n * K;
    Wt[i] = (half_t)W[(size_t)k * N + n];
  }
}

// ---------------------------------------------------------------------------
// f16 GEMM: C = act(A[M,K] @ Wt^T + bias), A row-major f16, Wt is N x K f16.
// 256 threads = 8 waves (4 row-slabs x 2 col-slabs); per wave 32x64 output
// = 2x4 fragments -> 8 v_wmma_f32_16x16x32_f16 per K-step.
// Tile staging is done by the Tensor Data Mover: wave 0 issues double-buffered
// tensor_load_to_lds for the NEXT A/B tiles while all waves run WMMAs on the
// current buffer, then drains TENSORcnt before the workgroup barrier.
// Epilogue writes f32 (C32) and/or f16 (C16).
// ---------------------------------------------------------------------------
__global__ void __launch_bounds__(256)
gemm_wmma_f16(const half_t* __restrict__ A, const half_t* __restrict__ Wt,
              const float* __restrict__ bias, float* __restrict__ C32,
              half_t* __restrict__ C16, int M, int N, int K, int relu)
{
  __shared__ alignas(16) half_t sA[2][BM][LDP];
  __shared__ alignas(16) half_t sB[2][BN][LDP];

  const int tid   = threadIdx.x;
  const int lane  = tid & 31;
  const int wave  = tid >> 5;
  const int waveM = wave & 3;    // 0..3 : 32-row slab
  const int waveN = wave >> 2;   // 0..1 : 64-col slab
  const int m0 = blockIdx.y * BM;
  const int n0 = blockIdx.x * BN;
  const int lrow = lane & 15;
  const int ka = (lane < 16) ? 0 : 8;   // A frag K-chunk base (ISA 7.12.2)
  const int kb = (lane < 16) ? 0 : 8;   // B frag K base
  const bool issuer = (wave == 0);      // uniform per wave: TDM ignores EXEC

  v8f acc[2][4];
  for (int i = 0; i < 2; ++i)
    for (int j = 0; j < 4; ++j)
      for (int e = 0; e < 8; ++e) acc[i][j][e] = 0.0f;

  // ---- prologue: DMA tile 0 into buffer 0 ----
  if (issuer) {
    tdm_load_tile_128x32(&A [(size_t)m0 * K], (unsigned int)(uintptr_t)&sA[0][0][0], K, M);
    tdm_load_tile_128x32(&Wt[(size_t)n0 * K], (unsigned int)(uintptr_t)&sB[0][0][0], K, N);
    __builtin_amdgcn_s_wait_tensorcnt(0);
  }
  __syncthreads();

  int buf = 0;
  for (int k0 = 0; k0 < K; k0 += BK) {
    const bool nxt = (k0 + BK) < K;

    // kick off DMA of the next tiles into the other buffer (overlaps WMMAs)
    if (issuer && nxt) {
      tdm_load_tile_128x32(&A [(size_t)m0 * K + k0 + BK],
                           (unsigned int)(uintptr_t)&sA[buf ^ 1][0][0], K, M);
      tdm_load_tile_128x32(&Wt[(size_t)n0 * K + k0 + BK],
                           (unsigned int)(uintptr_t)&sB[buf ^ 1][0][0], K, N);
    }

    // ---- fragments per ISA VGPR layout ----
    Frag16 af[2], bf[4];
    const int ar = waveM * 32 + lrow;
    for (int mf = 0; mf < 2; ++mf) {
      af[mf].h[0] = *(const v8h*)&sA[buf][ar + mf * 16][ka];
      af[mf].h[1] = *(const v8h*)&sA[buf][ar + mf * 16][ka + 16];
    }
    const int bc = waveN * 64 + lrow;
    for (int nf = 0; nf < 4; ++nf) {
      bf[nf].h[0] = *(const v8h*)&sB[buf][bc + nf * 16][kb];
      bf[nf].h[1] = *(const v8h*)&sB[buf][bc + nf * 16][kb + 16];
    }

    for (int mf = 0; mf < 2; ++mf)
      for (int nf = 0; nf < 4; ++nf)
        acc[mf][nf] = __builtin_amdgcn_wmma_f32_16x16x32_f16(
            false, af[mf].v, false, bf[nf].v, (short)0, acc[mf][nf], false, false);

    // drain the DMA (issuer only) before everyone flips buffers
    if (issuer && nxt) __builtin_amdgcn_s_wait_tensorcnt(0);
    __syncthreads();
    buf ^= 1;
  }

  // ---- epilogue: C/D layout (lane<16: M=e, N=lane; lane>=16: M=8+e) ----
  const int nl = lane & 15;
  const int mo = (lane >> 4) * 8;
  for (int mf = 0; mf < 2; ++mf)
    for (int nf = 0; nf < 4; ++nf)
      for (int e = 0; e < 8; ++e) {
        int grow = m0 + waveM * 32 + mf * 16 + mo + e;
        int gcol = n0 + waveN * 64 + nf * 16 + nl;
        float v = acc[mf][nf][e] + (bias ? bias[gcol] : 0.0f);
        if (relu) v = fmaxf(v, 0.0f);
        size_t off = (size_t)grow * N + gcol;
        if (C32) C32[off] = v;
        if (C16) C16[off] = (half_t)v;
      }
}

// ---------------------------------------------------------------------------
// Gumbel noise (hash-based uniform; deterministic per call)
// ---------------------------------------------------------------------------
__device__ __forceinline__ float gumbel_noise(uint32_t s) {
  s ^= s >> 16; s *= 0x7feb352dU; s ^= s >> 15; s *= 0x846ca68bU; s ^= s >> 16;
  float u = (float)(s >> 8) * (1.0f / 16777216.0f);
  u = fminf(fmaxf(u, 1e-12f), 1.0f - 1e-7f);
  return -logf(-logf(u));
}

// Sequential Gumbel-argmax selection (argmax(softmax((s+g)/tau)) == argmax(s+g)).
__global__ void __launch_bounds__(256)
select_tokens_kernel(const float* __restrict__ scores, int* __restrict__ sel_idx)
{
  __shared__ unsigned char avail[8192];
  __shared__ float rv[256];
  __shared__ int   ri[256];
  const int b = blockIdx.x, tid = threadIdx.x;
  for (int m = tid; m < 8192; m += 256) avail[m] = 1;
  __syncthreads();
  for (int k = 0; k < 128; ++k) {
    float best = -3.4e38f; int bi = 0;
    for (int m = tid; m < 8192; m += 256) {
      if (avail[m]) {
        uint32_t seed = ((uint32_t)(b * 128 + k) << 13) | (uint32_t)m;
        float v = scores[(size_t)(b * 8192 + m) * 128 + k] + gumbel_noise(seed);
        if (v > best) { best = v; bi = m; }
      }
    }
    rv[tid] = best; ri[tid] = bi; __syncthreads();
    for (int s = 128; s > 0; s >>= 1) {
      if (tid < s && rv[tid + s] > rv[tid]) { rv[tid] = rv[tid + s]; ri[tid] = ri[tid + s]; }
      __syncthreads();
    }
    if (tid == 0) { sel_idx[b * 128 + k] = ri[0]; avail[ri[0]] = 0; }
    __syncthreads();
  }
}

// Gather selected point features + coordinates (x,y,z,t).
__global__ void __launch_bounds__(256)
gather_selected_kernel(const float* __restrict__ pf, const float* __restrict__ coords,
                       const int* __restrict__ sel_idx, float* __restrict__ sel_feats,
                       float* __restrict__ sel_coords)
{
  const int t = blockIdx.x, tid = threadIdx.x;
  const int b = t >> 7;
  const int n = b * 8192 + sel_idx[t];
  for (int d = tid; d < 768; d += 256)
    sel_feats[(size_t)t * 768 + d] = pf[(size_t)n * 768 + d];
  if (tid < 4) sel_coords[t * 4 + tid] = coords[(size_t)n * 5 + 1 + tid];
}

// kNN (K=16): LDS-resident d^2 over 8192 pts, then 16 argmin sweeps.
__global__ void __launch_bounds__(256)
knn_kernel(const float* __restrict__ coords, const float* __restrict__ sel_coords,
           int* __restrict__ knn_idx)
{
  __shared__ float d2s[8192];
  __shared__ float rv[256];
  __shared__ int   ri[256];
  const int t = blockIdx.x, tid = threadIdx.x;
  const int b = t >> 7;
  const float sx = sel_coords[t * 4 + 0];
  const float sy = sel_coords[t * 4 + 1];
  const float sz = sel_coords[t * 4 + 2];
  for (int m = tid; m < 8192; m += 256) {
    const float* p = &coords[(size_t)(b * 8192 + m) * 5 + 1];
    float dx = p[0] - sx, dy = p[1] - sy, dz = p[2] - sz;
    d2s[m] = dx * dx + dy * dy + dz * dz;
  }
  __syncthreads();
  for (int j = 0; j < 16; ++j) {
    float best = 3.4e38f; int bi = 0;
    for (int m = tid; m < 8192; m += 256) {
      float v = d2s[m];
      if (v < best) { best = v; bi = m; }
    }
    rv[tid] = best; ri[tid] = bi; __syncthreads();
    for (int s = 128; s > 0; s >>= 1) {
      if (tid < s) {
        if (rv[tid + s] < rv[tid] || (rv[tid + s] == rv[tid] && ri[tid + s] < ri[tid])) {
          rv[tid] = rv[tid + s]; ri[tid] = ri[tid + s];
        }
      }
      __syncthreads();
    }
    if (tid == 0) { knn_idx[t * 16 + j] = ri[0]; d2s[ri[0]] = 3.4e38f; }
    __syncthreads();
  }
}

// Max-pool features of the 16 neighbors; emit f16 (next GEMM's A operand).
__global__ void __launch_bounds__(256)
pool_kernel(const float* __restrict__ pf, const int* __restrict__ knn_idx,
            half_t* __restrict__ pooled)
{
  const int t = blockIdx.x, tid = threadIdx.x;
  const int b = t >> 7;
  int idx[16];
  for (int j = 0; j < 16; ++j) idx[j] = b * 8192 + knn_idx[t * 16 + j];
  for (int d = tid; d < 768; d += 256) {
    float mx = -3.4e38f;
    for (int j = 0; j < 16; ++j) mx = fmaxf(mx, pf[(size_t)idx[j] * 768 + d]);
    pooled[(size_t)t * 768 + d] = (half_t)mx;
  }
}

__global__ void __launch_bounds__(256)
add_kernel(float* __restrict__ a, const float* __restrict__ b, int n)
{
  int i = blockIdx.x * 256 + threadIdx.x;
  if (i < n) a[i] += b[i];
}

// Per-batch bitonic argsort on time coordinate (index tie-break == stable sort).
__global__ void __launch_bounds__(128)
sort_kernel(const float* __restrict__ sel_coords, int* __restrict__ order)
{
  __shared__ float key[128];
  __shared__ int   val[128];
  const int b = blockIdx.x, tid = threadIdx.x;
  key[tid] = sel_coords[(b * 128 + tid) * 4 + 3];
  val[tid] = tid;
  __syncthreads();
  for (int ksz = 2; ksz <= 128; ksz <<= 1)
    for (int j = ksz >> 1; j > 0; j >>= 1) {
      int ixj = tid ^ j;
      if (ixj > tid) {
        bool up = ((tid & ksz) == 0);
        float ka = key[tid], kb = key[ixj];
        int   va = val[tid], vb = val[ixj];
        bool agb = (ka > kb) || (ka == kb && va > vb);
        if (up ? agb : !agb) {
          key[tid] = kb; key[ixj] = ka;
          val[tid] = vb; val[ixj] = va;
        }
      }
      __syncthreads();
    }
  order[b * 128 + tid] = val[tid];
}

// Emit tokens / centroids / mask in sorted order.
__global__ void __launch_bounds__(256)
writeout_kernel(const float* __restrict__ sel_feats, const float* __restrict__ sel_coords,
                const int* __restrict__ order, float* __restrict__ tokens,
                float* __restrict__ centroids, float* __restrict__ mask)
{
  const int t = blockIdx.x, tid = threadIdx.x;
  const int b = t >> 7, r = t & 127;
  const int src = b * 128 + order[b * 128 + r];
  for (int d = tid; d < 768; d += 256)
    tokens[(size_t)t * 768 + d] = sel_feats[(size_t)src * 768 + d];
  if (tid < 4) centroids[t * 4 + tid] = sel_coords[src * 4 + tid];
  if (tid == 0) mask[t] = 1.0f;
}

// ---------------------------------------------------------------------------
// Launcher
// ---------------------------------------------------------------------------
extern "C" void kernel_launch(void* const* d_in, const int* in_sizes, int n_in,
                              void* d_out, int out_size, void* d_ws, size_t ws_size,
                              hipStream_t stream)
{
  (void)in_sizes; (void)n_in; (void)out_size; (void)ws_size;

  const float* coords = (const float*)d_in[0];
  const float* feats  = (const float*)d_in[1];
  const float* W1 = (const float*)d_in[2];  const float* b1 = (const float*)d_in[3];
  const float* W2 = (const float*)d_in[4];  const float* b2 = (const float*)d_in[5];
  const float* W3 = (const float*)d_in[6];  const float* b3 = (const float*)d_in[7];
  const float* W4 = (const float*)d_in[8];  const float* b4 = (const float*)d_in[9];
  const float* sel_w = (const float*)d_in[10];
  const float* nW1 = (const float*)d_in[11]; const float* nb1 = (const float*)d_in[12];
  const float* nW2 = (const float*)d_in[13]; const float* nb2 = (const float*)d_in[14];

  // ---- workspace layout (256B-aligned chained allocs, with region reuse) ----
  char* p = (char*)d_ws;
  auto alloc = [&](size_t bytes) { char* r = p; p += (bytes + 255) & ~(size_t)255; return r; };

  float*  pf     = (float*) alloc((size_t)65536 * 768 * 4);   // f32 point feats (persistent)
  half_t* pf_h   = (half_t*)alloc((size_t)65536 * 768 * 2);   // f16 copy for scores GEMM
  char*   ha     =          alloc((size_t)65536 * 768 * 2);   // h1h/h3h -> small f32/f16 bufs
  char*   hb     =          alloc((size_t)65536 * 512 * 2);   // h2h -> scores + smalls

  half_t* feats_h = (half_t*)alloc((size_t)65536 * 32 * 2);
  half_t* W1t  = (half_t*)alloc((size_t)32  * 256 * 2);
  half_t* W2t  = (half_t*)alloc((size_t)256 * 512 * 2);
  half_t* W3t  = (half_t*)alloc((size_t)512 * 768 * 2);
  half_t* W4t  = (half_t*)alloc((size_t)768 * 768 * 2);
  half_t* selw_h = (half_t*)alloc((size_t)128 * 768 * 2);     // already N x K
  half_t* nW1t = (half_t*)alloc((size_t)768 * 768 * 2);
  half_t* nW2t = (half_t*)alloc((size_t)768 * 768 * 2);

  half_t* h1h = (half_t*)ha;        // 65536 x 256 f16
  half_t* h2h = (half_t*)hb;        // 65536 x 512 f16
  half_t* h3h = (half_t*)ha;        // 65536 x 768 f16
  float*  scores = (float*)hb;      // 65536 x 128 f32 (hb free after h3)

  // smalls in hb after scores
  char* hs = hb + (size_t)65536 * 128 * 4;
  int*   sel_idx    = (int*)hs;               hs += 8192;
  float* sel_coords = (float*)hs;             hs += 16384;
  int*   knn_idx    = (int*)hs;               hs += 65536;
  int*   order      = (int*)hs;               hs += 8192;

  // smalls in ha after pf is produced
  float*  sel_feats = (float*)ha;                         // 1024 x 768 f32
  half_t* pooled_h  = (half_t*)(ha + 3145728);            // 1024 x 768 f16
  half_t* agg1h     = (half_t*)(ha + 3145728 + 1572864);  // 1024 x 768 f16
  float*  agg2      = (float*)(ha + 3145728 + 3145728);   // 1024 x 768 f32

  const dim3 blk(256);
  auto cvt = [&](const float* s, half_t* d, int n) {
    convert_f16_kernel<<<(n + 255) / 256, blk, 0, stream>>>(s, d, n);
  };
  auto tr = [&](const float* Wm, half_t* Wo, int K, int N) {
    transpose_f16_kernel<<<(K * N + 255) / 256, blk, 0, stream>>>(Wm, Wo, K, N);
  };
  auto gemm = [&](const half_t* A, const half_t* Wt, const float* bi,
                  float* C32, half_t* C16, int M, int N, int K, int relu) {
    dim3 grid(N / BN, M / BM);
    gemm_wmma_f16<<<grid, blk, 0, stream>>>(A, Wt, bi, C32, C16, M, N, K, relu);
  };

  // ---- one-time f16 conversion of weights / input features ----
  cvt(feats, feats_h, 65536 * 32);
  cvt(sel_w, selw_h, 128 * 768);
  tr(W1, W1t, 32, 256);
  tr(W2, W2t, 256, 512);
  tr(W3, W3t, 512, 768);
  tr(W4, W4t, 768, 768);
  tr(nW1, nW1t, 768, 768);
  tr(nW2, nW2t, 768, 768);

  // ---- point MLP (dominant compute, all-WMMA, TDM-staged tiles) ----
  gemm(feats_h, W1t, b1, nullptr, h1h, 65536, 256,  32, 1);
  gemm(h1h,     W2t, b2, nullptr, h2h, 65536, 512, 256, 1);
  gemm(h2h,     W3t, b3, nullptr, h3h, 65536, 768, 512, 1);
  gemm(h3h,     W4t, b4, pf, pf_h,     65536, 768, 768, 0);

  // ---- selector scores: pf @ sel_w^T ----
  gemm(pf_h, selw_h, nullptr, scores, nullptr, 65536, 128, 768, 0);

  // ---- sequential token selection + gathers ----
  select_tokens_kernel<<<8, 256, 0, stream>>>(scores, sel_idx);
  gather_selected_kernel<<<1024, 256, 0, stream>>>(pf, coords, sel_idx, sel_feats, sel_coords);

  // ---- kNN + max-pool ----
  knn_kernel<<<1024, 256, 0, stream>>>(coords, sel_coords, knn_idx);
  pool_kernel<<<1024, 256, 0, stream>>>(pf, knn_idx, pooled_h);

  // ---- neighborhood MLP (WMMA) and residual add ----
  gemm(pooled_h, nW1t, nb1, nullptr, agg1h, 1024, 768, 768, 1);
  gemm(agg1h,    nW2t, nb2, agg2, nullptr,  1024, 768, 768, 0);
  add_kernel<<<(1024 * 768) / 256, 256, 0, stream>>>(sel_feats, agg2, 1024 * 768);

  // ---- sort by time and write outputs: tokens | centroids | mask ----
  sort_kernel<<<8, 128, 0, stream>>>(sel_coords, order);
  float* out = (float*)d_out;
  writeout_kernel<<<1024, 256, 0, stream>>>(sel_feats, sel_coords, order,
                                            out, out + 8 * 128 * 768,
                                            out + 8 * 128 * 768 + 8 * 128 * 4);
}
